// SiMaC_27711128994587
// MI455X (gfx1250) — compile-verified, matrix operand order checked by reading, not run on hardware
//
#include <hip/hip_runtime.h>
#include <hip/hip_fp16.h>

#define B_DIM   2048
#define C_DIM   1024
#define F_DIM   128
#define N_USERS 100000
#define EPSILON 0.1f
#define ALPHA   0.5f
#define N_ITER  10

typedef __attribute__((ext_vector_type(16))) _Float16 v16h;
typedef __attribute__((ext_vector_type(8)))  float    v8f;

// Load 16 f16 elements of a WMMA A/B fragment from a contiguous f32 row.
// Per the CDNA5 16-bit A-matrix 16x32 layout: element i of the v16h maps to
// K = kbase + kb + ((i&8)?16:0) + (i&7), with kb = (lane>=16)?8:0.
// That is two contiguous 8-float runs -> two float4 pairs.
__device__ __forceinline__ v16h load_frag_f32(const float* __restrict__ row,
                                              int kbase, int kb) {
  const float4 f0 = *(const float4*)(row + kbase + kb);
  const float4 f1 = *(const float4*)(row + kbase + kb + 4);
  const float4 f2 = *(const float4*)(row + kbase + 16 + kb);
  const float4 f3 = *(const float4*)(row + kbase + 16 + kb + 4);
  v16h r;
  r[0]  = (_Float16)f0.x; r[1]  = (_Float16)f0.y;
  r[2]  = (_Float16)f0.z; r[3]  = (_Float16)f0.w;
  r[4]  = (_Float16)f1.x; r[5]  = (_Float16)f1.y;
  r[6]  = (_Float16)f1.z; r[7]  = (_Float16)f1.w;
  r[8]  = (_Float16)f2.x; r[9]  = (_Float16)f2.y;
  r[10] = (_Float16)f2.z; r[11] = (_Float16)f2.w;
  r[12] = (_Float16)f3.x; r[13] = (_Float16)f3.y;
  r[14] = (_Float16)f3.z; r[15] = (_Float16)f3.w;
  return r;
}

// ---- init workspace: sumD accumulator, u = ones(B), v = ones(C) ----
__global__ __launch_bounds__(1024) void init_ws(float* sumD, float* u, float* v) {
  const int t = threadIdx.x;
  if (t == 0) sumD[0] = 0.0f;
  if (t < C_DIM) v[t] = 1.0f;
  u[t] = 1.0f;
  u[t + 1024] = 1.0f;
}

// ---- sum(D) via block partials + one atomic per block ----
__global__ __launch_bounds__(256) void reduce_sumD(const float* __restrict__ D,
                                                   float* __restrict__ sumD) {
  __shared__ float sm[256];
  float s = 0.0f;
  for (size_t i = (size_t)blockIdx.x * 256 + threadIdx.x;
       i < (size_t)B_DIM * C_DIM; i += (size_t)256 * 256)
    s += D[i];
  sm[threadIdx.x] = s;
  __syncthreads();
  for (int off = 128; off > 0; off >>= 1) {
    if (threadIdx.x < off) sm[threadIdx.x] += sm[threadIdx.x + off];
    __syncthreads();
  }
  if (threadIdx.x == 0) atomicAdd(sumD, sm[0]);
}

// ---- WMMA GEMM (M = UE x IE^T) fused with gather + K = exp(affinity/eps) ----
// One block computes 16 rows of M (16x1024, 64KB LDS), 8 waves x 8 col-tiles.
__global__ __launch_bounds__(256) void gemm_build_K(
    const int* __restrict__ users, const int* __restrict__ items,
    const float* __restrict__ D, const float* __restrict__ item_emb,
    const float* __restrict__ user_emb, const float* __restrict__ sumD,
    float* __restrict__ K) {
  __shared__ float Mtile[16 * C_DIM];             // 64 KB

  const int tid   = threadIdx.x;
  const int wave  = tid >> 5;                     // 0..7
  const int lane  = tid & 31;
  const int lrow  = lane & 15;                    // row/col within 16-tile
  const int kb    = (lane >> 4) << 3;             // 0 or 8 (K sub-offset)
  const int moff  = (lane >> 4) << 3;             // C/D row sub-offset
  const int brow0 = blockIdx.x * 16;

  // A fragments (user embeddings), shared across all column tiles of this wave
  const int   grow = brow0 + lrow;
  const int   uidx = users[grow];
  const float* arow = user_emb + (size_t)uidx * F_DIM;
  const v16h a0 = load_frag_f32(arow, 0,  kb);
  const v16h a1 = load_frag_f32(arow, 32, kb);
  const v16h a2 = load_frag_f32(arow, 64, kb);
  const v16h a3 = load_frag_f32(arow, 96, kb);

  for (int t = 0; t < 8; ++t) {
    const int ct   = wave * 8 + t;                // column tile 0..63
    const int ncol = ct * 16 + lrow;              // this lane's N column
    const float* brow = item_emb + (size_t)ncol * F_DIM;

    v8f acc = {0.f, 0.f, 0.f, 0.f, 0.f, 0.f, 0.f, 0.f};
    v16h b0 = load_frag_f32(brow, 0, kb);
    acc = __builtin_amdgcn_wmma_f32_16x16x32_f16(false, a0, false, b0,
                                                 (short)0, acc, false, false);
    v16h b1 = load_frag_f32(brow, 32, kb);
    acc = __builtin_amdgcn_wmma_f32_16x16x32_f16(false, a1, false, b1,
                                                 (short)0, acc, false, false);
    v16h b2 = load_frag_f32(brow, 64, kb);
    acc = __builtin_amdgcn_wmma_f32_16x16x32_f16(false, a2, false, b2,
                                                 (short)0, acc, false, false);
    v16h b3 = load_frag_f32(brow, 96, kb);
    acc = __builtin_amdgcn_wmma_f32_16x16x32_f16(false, a3, false, b3,
                                                 (short)0, acc, false, false);

    // D layout: VGPR r -> row (r + moff), lane&15 -> column
    #pragma unroll
    for (int r = 0; r < 8; ++r)
      Mtile[(r + moff) * C_DIM + ncol] = acc[r];
  }
  __syncthreads();

  // K[b,c] = exp( ((1-a)*M[b,items[b,c]] - a*D[b,c]*(B*C)/sum(D)) / eps )
  const float bc_over_sum = (float)((size_t)B_DIM * C_DIM) / sumD[0];
  const float w_dot = (1.0f - ALPHA) / EPSILON;   // 5.0
  const float w_d   = (ALPHA / EPSILON) * bc_over_sum;
  for (int i = tid; i < 16 * C_DIM; i += 256) {
    const int lr = i >> 10;
    const int c  = i & (C_DIM - 1);
    const size_t gidx = (size_t)(brow0 + lr) * C_DIM + c;
    const int j = items[gidx];
    const float dot = Mtile[lr * C_DIM + j];
    K[gidx] = __expf(w_dot * dot - w_d * D[gidx]);
  }
}

// ---- Sinkhorn row step: u[b] = 1 / sum_c K[b,c] * v[c]  (one wave per row) ----
__global__ __launch_bounds__(256) void row_step(const float* __restrict__ K,
                                                const float* __restrict__ v,
                                                float* __restrict__ u) {
  const int wave = threadIdx.x >> 5;
  const int lane = threadIdx.x & 31;
  const int row  = blockIdx.x * 8 + wave;
  const float* kr = K + (size_t)row * C_DIM;
  float s = 0.0f;
  #pragma unroll
  for (int i = 0; i < 8; ++i) {
    const int c = (i * 32 + lane) * 4;
    const float4 kk = *(const float4*)(kr + c);
    const float4 vv = *(const float4*)(v + c);
    s += kk.x * vv.x + kk.y * vv.y + kk.z * vv.z + kk.w * vv.w;
  }
  #pragma unroll
  for (int off = 16; off > 0; off >>= 1)
    s += __shfl_xor(s, off, 32);
  if (lane == 0) u[row] = 1.0f / s;
}

// ---- Sinkhorn col step: v[c] = caps[c]*scale / sum_b K[b,c] * u[b] ----
__global__ __launch_bounds__(256) void col_step(const float* __restrict__ K,
                                                const float* __restrict__ u,
                                                const float* __restrict__ caps,
                                                float* __restrict__ v) {
  const int c = blockIdx.x * 256 + threadIdx.x;
  float s = 0.0f;
  #pragma unroll 8
  for (int b = 0; b < B_DIM; ++b)
    s += K[(size_t)b * C_DIM + c] * u[b];        // u[b] uniform -> broadcast
  const float scaling = (float)B_DIM / (float)N_USERS;
  v[c] = caps[c] * scaling / s;
}

// ---- P = K * outer(u, v), in place on d_out ----
__global__ __launch_bounds__(256) void final_scale(float* __restrict__ P,
                                                   const float* __restrict__ u,
                                                   const float* __restrict__ v) {
  const size_t i4 = (size_t)blockIdx.x * 256 + threadIdx.x;  // float4 index
  float4* P4 = (float4*)P;
  const int b  = (int)(i4 >> 8);                 // 256 float4 per row
  const int c0 = (int)(i4 & 255) * 4;
  float4 p = P4[i4];
  const float ub = u[b];
  p.x *= ub * v[c0 + 0];
  p.y *= ub * v[c0 + 1];
  p.z *= ub * v[c0 + 2];
  p.w *= ub * v[c0 + 3];
  P4[i4] = p;
}

extern "C" void kernel_launch(void* const* d_in, const int* in_sizes, int n_in,
                              void* d_out, int out_size, void* d_ws, size_t ws_size,
                              hipStream_t stream) {
  (void)in_sizes; (void)n_in; (void)out_size; (void)ws_size;
  const int*   users    = (const int*)d_in[0];
  const int*   items    = (const int*)d_in[1];
  const float* D        = (const float*)d_in[2];
  const float* caps     = (const float*)d_in[3];
  const float* item_emb = (const float*)d_in[4];
  const float* user_emb = (const float*)d_in[5];

  float* K  = (float*)d_out;                     // K lives in d_out (8 MB)
  float* ws = (float*)d_ws;
  float* sumD = ws;                              // 1 float
  float* u    = ws + 64;                         // 2048 floats, 256B-aligned
  float* v    = ws + 64 + B_DIM;                 // 1024 floats

  init_ws<<<1, 1024, 0, stream>>>(sumD, u, v);
  reduce_sumD<<<256, 256, 0, stream>>>(D, sumD);
  gemm_build_K<<<B_DIM / 16, 256, 0, stream>>>(users, items, D, item_emb,
                                               user_emb, sumD, K);
  for (int it = 0; it < N_ITER; ++it) {
    row_step<<<B_DIM / 8, 256, 0, stream>>>(K, v, u);
    col_step<<<C_DIM / 256, 256, 0, stream>>>(K, u, caps, v);
  }
  final_scale<<<((size_t)B_DIM * C_DIM / 4) / 256, 256, 0, stream>>>(K, u, v);
}
